// MultiHeadSelfAttention_79070347919747
// MI455X (gfx1250) — matrix operand order repaired and checked by reading, not compiled, not run on hardware
//
#include <hip/hip_runtime.h>
#include <hip/hip_bf16.h>

// MI455X (gfx1250) wave32 WMMA implementation.
// Pipeline (all f16 WMMA with f32 accumulation):
//   0a. convert Wq/Wk/Wv/Wo to f16
//   0b. transpose+convert x [b,c,p] -> xT [b,p,c] (f16) so GEMM B-frags are contiguous
//   1.  Q/K/V = W @ X  (M=512,N=1024,K=256 per batch), Q/K stored [bh][n][64], V stored [bh][64][n]
//       Q is pre-scaled by 64^-0.5 * log2(e) so attention logits are in base-2 space.
//   2.  flash attention per (b,h): online softmax (exp2-space), O stored [b][n][512] f16
//   3.  y = Wo @ O + bo  -> d_out (f32, pre-BN)
//   4.  BN stats per channel, 5. BN apply in place.
// Workspace use (~37.01 MiB): xT 4M | Wq/Wk/Wv/Wo f16 1M | Q 8M | K 8M | V 8M | O 8M | stats 2K

typedef __attribute__((ext_vector_type(16))) _Float16 v16h;
typedef __attribute__((ext_vector_type(8)))  _Float16 v8h;
typedef __attribute__((ext_vector_type(8)))  float    v8f;

#define QSCALE 0.180336880094680073974f   // 64^-0.5 * log2(e)

__device__ __forceinline__ v16h load_v16h(const _Float16* p) {
  v16h r;
  *(v8h*)&r       = *(const v8h*)p;
  *((v8h*)&r + 1) = *(const v8h*)(p + 8);
  return r;
}
// A-fragment: two 8-half chunks at independent offsets (CDNA5 16-bit A layout)
__device__ __forceinline__ v16h load_afrag(const _Float16* row, int off0, int off1) {
  v16h r;
  *(v8h*)&r       = *(const v8h*)(row + off0);
  *((v8h*)&r + 1) = *(const v8h*)(row + off1);
  return r;
}
__device__ __forceinline__ v8f wmma_f16(v16h a, v16h b, v8f c) {
  return __builtin_amdgcn_wmma_f32_16x16x32_f16(false, a, false, b, (short)0, c, false, false);
}

// ---------------- 0a: f32 -> f16 convert ----------------
__global__ void cvt_f16_kernel(const float* __restrict__ src, _Float16* __restrict__ dst, int n) {
  int i = blockIdx.x * 256 + threadIdx.x;
  if (i < n) dst[i] = (_Float16)src[i];
}

// ---------------- 0b: x [b][256][1024] f32 -> xT [b][1024][256] f16 ----------------
__global__ __launch_bounds__(256) void xt_kernel(const float* __restrict__ x,
                                                 _Float16* __restrict__ xt) {
  __shared__ _Float16 t[32][33];
  const int tx = threadIdx.x & 31, ty = threadIdx.x >> 5;
  const int p0 = blockIdx.x * 32, c0 = blockIdx.y * 32, b = blockIdx.z;
#pragma unroll
  for (int j = 0; j < 4; ++j) {
    int c = c0 + ty + j * 8;
    t[ty + j * 8][tx] = (_Float16)x[((size_t)b * 256 + c) * 1024 + p0 + tx];
  }
  __syncthreads();
#pragma unroll
  for (int j = 0; j < 4; ++j) {
    int p = p0 + ty + j * 8;
    xt[((size_t)b * 1024 + p) * 256 + c0 + tx] = t[tx][ty + j * 8];
  }
}

// ---------------- 1: QKV projection GEMM ----------------
// out[o][p] = sum_c W[o][c] * xT[b][p][c].
// Per wave: 32(M) x 64(N) tile -> 8 accumulators, 8 WMMAs per k-step, each B-frag used twice.
// vlayout 0: dst[bh][p][d] (Q,K, scaled by oscale);  vlayout 1: dst[bh][d][p] (V)
__global__ __launch_bounds__(256) void qkv_proj_kernel(
    const _Float16* __restrict__ Wh,  // [512][256]
    const _Float16* __restrict__ Xt,  // [B][1024][256]
    _Float16* __restrict__ dst, int vlayout, float oscale) {
  const int tid = threadIdx.x;
  const int wave = tid >> 5, lane = tid & 31;
  const int half = lane >> 4, ln = lane & 15;
  const int b  = blockIdx.z;
  const int wm = wave & 3, wn = wave >> 2;
  const int o0 = blockIdx.x * 128 + wm * 32;   // two 16-row m-tiles
  const int p0 = blockIdx.y * 128 + wn * 64;   // four 16-col n-tiles

  const _Float16* wrow0 = Wh + (size_t)(o0 + ln) * 256 + half * 8;
  const _Float16* wrow1 = wrow0 + (size_t)16 * 256;
  const _Float16* xrow  = Xt + ((size_t)b * 1024 + p0) * 256 + half * 8;

  v8f acc[2][4] = {};
#pragma unroll
  for (int t = 0; t < 8; ++t) {
    v16h a0 = load_afrag(wrow0 + t * 32, 0, 16);
    v16h a1 = load_afrag(wrow1 + t * 32, 0, 16);
    v16h bf[4];
#pragma unroll
    for (int j = 0; j < 4; ++j)
      bf[j] = load_afrag(xrow + (size_t)(ln + 16 * j) * 256 + t * 32, 0, 16);
#pragma unroll
    for (int j = 0; j < 4; ++j) {
      acc[0][j] = wmma_f16(a0, bf[j], acc[0][j]);
      acc[1][j] = wmma_f16(a1, bf[j], acc[1][j]);
    }
  }

#pragma unroll
  for (int mi = 0; mi < 2; ++mi) {
    const int om = o0 + mi * 16;
    const int hd = om >> 6;
    const int db = (om & 63) + 8 * half;       // 8 consecutive d values per lane
    const size_t bh = (size_t)b * 8 + hd;
#pragma unroll
    for (int j = 0; j < 4; ++j) {
      const v8f cf = acc[mi][j];
      const int p = p0 + j * 16 + ln;
      if (vlayout == 0) {
        v8h pk;
#pragma unroll
        for (int r = 0; r < 8; ++r) pk[r] = (_Float16)(cf[r] * oscale);
        *(v8h*)(dst + (bh * 1024 + p) * 64 + db) = pk;
      } else {
#pragma unroll
        for (int r = 0; r < 8; ++r)
          dst[(bh * 64 + db + r) * 1024 + p] = (_Float16)cf[r];
      }
    }
  }
}

// ---------------- 2: fused flash attention ----------------
// grid (64 bh, 8 qblocks), 8 waves x 16 query rows each, key blocks of 32.
// Q was pre-scaled so logits are base-2: softmax = exp2(z - m) / sum.
__global__ __launch_bounds__(256) void attn_kernel(
    const _Float16* __restrict__ Q,  // [bh][n][64]  (pre-scaled)
    const _Float16* __restrict__ K,  // [bh][n][64]
    const _Float16* __restrict__ V,  // [bh][64][n]
    _Float16* __restrict__ O) {      // [b][n][512]
  __shared__ _Float16 lds_p[8][16 * 40];  // padded rows: conflict-free ds_read_b128
  const int tid = threadIdx.x;
  const int wave = tid >> 5, lane = tid & 31;
  const int half = lane >> 4, ln = lane & 15;
  const int bh = blockIdx.x;
  const int q0 = blockIdx.y * 128 + wave * 16;
  const int b = bh >> 3, h = bh & 7;

  const _Float16* qrow = Q + ((size_t)bh * 1024 + q0 + ln) * 64;
  const v16h aq0 = load_afrag(qrow,      half * 16, half * 16 + 8);  // d 0..31
  const v16h aq1 = load_afrag(qrow + 32, half * 16, half * 16 + 8);  // d 32..63

  v8f o0 = {}, o1 = {}, o2 = {}, o3 = {};
  float m_i[8], l_i[8];
#pragma unroll
  for (int r = 0; r < 8; ++r) { m_i[r] = -3.0e38f; l_i[r] = 0.0f; }

  _Float16* pbuf = &lds_p[wave][0];
  const _Float16* kwav = K + (size_t)bh * 1024 * 64 + half * 16;
  const _Float16* vwav = V + (size_t)bh * 64 * 1024 + half * 16;

  for (int kb = 0; kb < 1024; kb += 32) {
    if (kb + 32 < 1024) {  // prefetch next key block (L2-resident) -> global_prefetch_b8
      __builtin_prefetch(kwav + (size_t)(kb + 32 + ln) * 64, 0, 3);
      __builtin_prefetch(vwav + (size_t)(ln * 4) * 1024 + kb + 32, 0, 3);
    }
    const _Float16* kbase = kwav + (size_t)kb * 64;
    v16h bk00 = load_v16h(kbase + (size_t)(ln)      * 64);
    v16h bk01 = load_v16h(kbase + (size_t)(ln)      * 64 + 32);
    v16h bk10 = load_v16h(kbase + (size_t)(ln + 16) * 64);
    v16h bk11 = load_v16h(kbase + (size_t)(ln + 16) * 64 + 32);

    v8f z = {};
    v8f s0 = wmma_f16(aq0, bk00, z); s0 = wmma_f16(aq1, bk01, s0);  // keys kb+0..15
    v8f s1 = wmma_f16(aq0, bk10, z); s1 = wmma_f16(aq1, bk11, s1);  // keys kb+16..31

    float corr[8];
#pragma unroll
    for (int r = 0; r < 8; ++r) {
      float v = fmaxf(s0[r], s1[r]);
      v = fmaxf(v, __shfl_xor(v, 1, 16));
      v = fmaxf(v, __shfl_xor(v, 2, 16));
      v = fmaxf(v, __shfl_xor(v, 4, 16));
      v = fmaxf(v, __shfl_xor(v, 8, 16));
      float mn = fmaxf(m_i[r], v);
      corr[r] = exp2f(m_i[r] - mn);
      m_i[r] = mn;
    }
#pragma unroll
    for (int r = 0; r < 8; ++r) {
      float p0v = exp2f(s0[r] - m_i[r]);
      float p1v = exp2f(s1[r] - m_i[r]);
      float rs = p0v + p1v;
      rs += __shfl_xor(rs, 1, 16);
      rs += __shfl_xor(rs, 2, 16);
      rs += __shfl_xor(rs, 4, 16);
      rs += __shfl_xor(rs, 8, 16);
      l_i[r] = l_i[r] * corr[r] + rs;
      o0[r] *= corr[r]; o1[r] *= corr[r]; o2[r] *= corr[r]; o3[r] *= corr[r];
      // stage P (C-frag layout -> row-major LDS) for reuse as A-frag
      pbuf[(r + 8 * half) * 40 + ln]      = (_Float16)p0v;
      pbuf[(r + 8 * half) * 40 + 16 + ln] = (_Float16)p1v;
    }
    asm volatile("s_wait_dscnt 0" ::: "memory");   // cross-lane LDS RAW inside wave
    v16h pa = load_afrag(pbuf + ln * 40, half * 16, half * 16 + 8);

    const _Float16* vbase = vwav + kb;
    o0 = wmma_f16(pa, load_v16h(vbase + (size_t)(ln)      * 1024), o0);
    o1 = wmma_f16(pa, load_v16h(vbase + (size_t)(ln + 16) * 1024), o1);
    o2 = wmma_f16(pa, load_v16h(vbase + (size_t)(ln + 32) * 1024), o2);
    o3 = wmma_f16(pa, load_v16h(vbase + (size_t)(ln + 48) * 1024), o3);
  }

#pragma unroll
  for (int r = 0; r < 8; ++r) {
    float inv = 1.0f / l_i[r];
    size_t row = ((size_t)b * 1024 + q0 + r + 8 * half) * 512 + h * 64;
    O[row + ln]      = (_Float16)(o0[r] * inv);
    O[row + 16 + ln] = (_Float16)(o1[r] * inv);
    O[row + 32 + ln] = (_Float16)(o2[r] * inv);
    O[row + 48 + ln] = (_Float16)(o3[r] * inv);
  }
}

// ---------------- 3: output projection + bias ----------------
// Per wave: 32(M) x 64(N) tile, K=512.
__global__ __launch_bounds__(256) void out_proj_kernel(
    const _Float16* __restrict__ Woh,  // [256][512]
    const _Float16* __restrict__ O,    // [B][1024][512]
    const float* __restrict__ bo,
    float* __restrict__ y) {           // [B][256][1024] pre-BN
  const int tid = threadIdx.x;
  const int wave = tid >> 5, lane = tid & 31;
  const int half = lane >> 4, ln = lane & 15;
  const int b  = blockIdx.z;
  const int wm = wave & 3, wn = wave >> 2;
  const int c0 = blockIdx.x * 128 + wm * 32;
  const int p0 = blockIdx.y * 128 + wn * 64;

  const _Float16* wrow0 = Woh + (size_t)(c0 + ln) * 512 + half * 8;
  const _Float16* wrow1 = wrow0 + (size_t)16 * 512;
  const _Float16* orow  = O + ((size_t)b * 1024 + p0) * 512 + half * 8;

  v8f acc[2][4] = {};
#pragma unroll
  for (int t = 0; t < 16; ++t) {
    v16h a0 = load_afrag(wrow0 + t * 32, 0, 16);
    v16h a1 = load_afrag(wrow1 + t * 32, 0, 16);
    v16h bf[4];
#pragma unroll
    for (int j = 0; j < 4; ++j)
      bf[j] = load_afrag(orow + (size_t)(ln + 16 * j) * 512 + t * 32, 0, 16);
#pragma unroll
    for (int j = 0; j < 4; ++j) {
      acc[0][j] = wmma_f16(a0, bf[j], acc[0][j]);
      acc[1][j] = wmma_f16(a1, bf[j], acc[1][j]);
    }
  }

#pragma unroll
  for (int mi = 0; mi < 2; ++mi) {
#pragma unroll
    for (int j = 0; j < 4; ++j) {
      const v8f cf = acc[mi][j];
      const int p = p0 + j * 16 + ln;
#pragma unroll
      for (int r = 0; r < 8; ++r) {
        int c = c0 + mi * 16 + r + 8 * half;
        y[((size_t)b * 256 + c) * 1024 + p] = cf[r] + bo[c];
      }
    }
  }
}

// ---------------- 4: BN stats (one block per channel) ----------------
__global__ __launch_bounds__(256) void bnstats_kernel(const float* __restrict__ y,
                                                      float* __restrict__ stats) {
  __shared__ float ssum[256], ssq[256];
  const int c = blockIdx.x, tid = threadIdx.x;
  float s = 0.0f, q = 0.0f;
  for (int i = tid; i < 8192; i += 256) {
    int b = i >> 10, p = i & 1023;
    float v = y[((size_t)b * 256 + c) * 1024 + p];
    s += v; q += v * v;
  }
  ssum[tid] = s; ssq[tid] = q;
  __syncthreads();
  for (int off = 128; off > 0; off >>= 1) {
    if (tid < off) { ssum[tid] += ssum[tid + off]; ssq[tid] += ssq[tid + off]; }
    __syncthreads();
  }
  if (tid == 0) {
    float mean = ssum[0] * (1.0f / 8192.0f);
    float var  = ssq[0]  * (1.0f / 8192.0f) - mean * mean;
    stats[c * 2]     = mean;
    stats[c * 2 + 1] = rsqrtf(var + 1e-5f);
  }
}

// ---------------- 5: BN apply (in place) ----------------
__global__ void bnapply_kernel(float* __restrict__ y, const float* __restrict__ stats,
                               const float* __restrict__ gamma,
                               const float* __restrict__ beta, int n) {
  int i = blockIdx.x * 256 + threadIdx.x;
  if (i >= n) return;
  int c = (i >> 10) & 255;
  y[i] = (y[i] - stats[c * 2]) * stats[c * 2 + 1] * gamma[c] + beta[c];
}

extern "C" void kernel_launch(void* const* d_in, const int* in_sizes, int n_in,
                              void* d_out, int out_size, void* d_ws, size_t ws_size,
                              hipStream_t stream) {
  const float* x     = (const float*)d_in[0];
  const float* Wq    = (const float*)d_in[1];
  const float* Wk    = (const float*)d_in[2];
  const float* Wv    = (const float*)d_in[3];
  const float* Wo    = (const float*)d_in[4];
  const float* bo    = (const float*)d_in[5];
  const float* gamma = (const float*)d_in[6];
  const float* beta  = (const float*)d_in[7];

  char* ws = (char*)d_ws;
  _Float16* xt  = (_Float16*)(ws);                                   //  4 MiB
  _Float16* wqh = (_Float16*)(ws + ((size_t)4 << 20));               //  256 KiB
  _Float16* wkh = (_Float16*)(ws + ((size_t)4 << 20) + (256 << 10));
  _Float16* wvh = (_Float16*)(ws + ((size_t)4 << 20) + (512 << 10));
  _Float16* woh = (_Float16*)(ws + ((size_t)4 << 20) + (768 << 10));
  _Float16* Qw  = (_Float16*)(ws + ((size_t)5  << 20));              //  8 MiB
  _Float16* Kw  = (_Float16*)(ws + ((size_t)13 << 20));              //  8 MiB
  _Float16* Vw  = (_Float16*)(ws + ((size_t)21 << 20));              //  8 MiB
  _Float16* Ow  = (_Float16*)(ws + ((size_t)29 << 20));              //  8 MiB
  float*    st  = (float*)   (ws + ((size_t)37 << 20));              //  2 KiB
  float* y = (float*)d_out;

  cvt_f16_kernel<<<512, 256, 0, stream>>>(Wq, wqh, 131072);
  cvt_f16_kernel<<<512, 256, 0, stream>>>(Wk, wkh, 131072);
  cvt_f16_kernel<<<512, 256, 0, stream>>>(Wv, wvh, 131072);
  cvt_f16_kernel<<<512, 256, 0, stream>>>(Wo, woh, 131072);
  xt_kernel<<<dim3(32, 8, 8), 256, 0, stream>>>(x, xt);

  qkv_proj_kernel<<<dim3(4, 8, 8), 256, 0, stream>>>(wqh, xt, Qw, 0, QSCALE);
  qkv_proj_kernel<<<dim3(4, 8, 8), 256, 0, stream>>>(wkh, xt, Kw, 0, 1.0f);
  qkv_proj_kernel<<<dim3(4, 8, 8), 256, 0, stream>>>(wvh, xt, Vw, 1, 1.0f);

  attn_kernel<<<dim3(64, 8), 256, 0, stream>>>(Qw, Kw, Vw, Ow);

  out_proj_kernel<<<dim3(2, 8, 8), 256, 0, stream>>>(woh, Ow, bo, y);

  bnstats_kernel<<<256, 256, 0, stream>>>(y, st);
  bnapply_kernel<<<8192, 256, 0, stream>>>(y, st, gamma, beta, out_size);
}
